// HeteroGAT_pyg_17119739641950
// MI455X (gfx1250) — matrix operand order repaired and checked by reading, not compiled, not run on hardware
//
#include <hip/hip_runtime.h>
#include <hip/hip_bf16.h>

// ---------------------------------------------------------------------------
// HeteroGAT forward, MI455X (gfx1250, wave32).
// Only the protein(ppi) branch reaches the output (hc2 is dead in the ref).
//   H1 = xp @ Wa[2]                               (WMMA f16->f32 GEMM)
//   GAT softmax-aggregate over ei_ppi + selfloops (D=256) -> hp = relu(+bias)
//   H2 = hp @ Wb[2]                               (WMMA GEMM)
//   GAT softmax-aggregate again (D=128)           -> hp2 = +bias
//   s = hp2 . Wout[:128], t = hp2 . Wout[128:],   out = sigmoid(s[m0]+t[m1]+b)
// ---------------------------------------------------------------------------

typedef __attribute__((ext_vector_type(16))) _Float16 v16h;
typedef __attribute__((ext_vector_type(8)))  float    v8f;

#define FDIM 128
#define HDIM 256

// ---------------------------------------------------------------------------
// WMMA GEMM: C[M,N] = A[M,K] @ B[K,N]; fp32 in/out, f16 multiply, f32 acc.
// K is a compile-time template arg (128 / 256) so the K loop fully unrolls
// into back-to-back v_wmma ops.  Block = 256 threads = 8 waves; each wave
// owns one 16x16 C tile (8 M-tiles per block, one N-tile per blockIdx.y).
// The 16-column B panel is staged once per block into LDS as f16 in
// column-major order, so each lane's B fragment is two contiguous 16-byte
// LDS reads.  M % 16 == 0; trailing waves exit after the barrier on a
// wave-uniform branch so EXEC is all-ones for every WMMA.
// ---------------------------------------------------------------------------
template <int K>
__global__ __launch_bounds__(256)
void gemm_wmma_f16(const float* __restrict__ A, const float* __restrict__ B,
                   float* __restrict__ C, int M, int N)
{
    constexpr int KP = K + 16;                 // padded column stride (halves)
    __shared__ _Float16 bt[16 * KP];           // B panel, column-major

    const int tn  = blockIdx.y;                // N/16 tile
    const int tid = threadIdx.x;

    // ---- cooperative B panel load: B[k][tn*16+col] -> bt[col*KP + k] ----
    #pragma unroll
    for (int idx = tid; idx < 16 * K; idx += 256) {
        int col = idx & 15;
        int k   = idx >> 4;
        bt[col * KP + k] = (_Float16)B[(size_t)k * N + tn * 16 + col];
    }
    __syncthreads();

    const int lane  = tid & 31;
    const int hsel  = lane >> 4;
    const int l15   = lane & 15;
    const int tileM = blockIdx.x * 8 + (tid >> 5);  // one M-tile per wave
    if (tileM * 16 >= M) return;                    // wave-uniform exit
    const int row = tileM * 16 + l15;

    v8f acc = {};
    #pragma unroll
    for (int k0 = 0; k0 < K; k0 += 32) {
        // A 16x32 f16 fragment: lane L -> row L&15; elems j=0..7 are
        // K = k0 + (L>>4)*8 + j ; elems j=8..15 add 16.
        v16h a, b;
        const float* Ap = A + (size_t)row * K + k0 + hsel * 8;
        #pragma unroll
        for (int j = 0; j < 8; ++j)  a[j]     = (_Float16)Ap[j];
        #pragma unroll
        for (int j = 0; j < 8; ++j)  a[8 + j] = (_Float16)Ap[16 + j];
        // B 32x16 fragment: lane L -> col L&15; elems j=0..7 are
        // K = k0 + (L>>4)*8 + j ; elems j=8..15 add 16.
        const _Float16* Bp = &bt[l15 * KP + k0 + hsel * 8];
        #pragma unroll
        for (int j = 0; j < 8; ++j)  b[j]     = Bp[j];
        #pragma unroll
        for (int j = 0; j < 8; ++j)  b[8 + j] = Bp[16 + j];

        acc = __builtin_amdgcn_wmma_f32_16x16x32_f16(
                  false, a, false, b, (short)0, acc, false, false);
    }
    // D 16x16 f32 layout: lane L, reg r -> row (L>>4)*8+r, col L&15.
    float* Cp = C + (size_t)(tileM * 16 + hsel * 8) * N + tn * 16 + l15;
    #pragma unroll
    for (int r = 0; r < 8; ++r) Cp[(size_t)r * N] = acc[r];
}

// ---------------------------------------------------------------------------
// Wave-per-node double dot product: o1[i] = H[i,:].v1, o2[i] = H[i,:].v2
// ---------------------------------------------------------------------------
__device__ __forceinline__ float wave_sum(float v)
{
    #pragma unroll
    for (int off = 16; off > 0; off >>= 1) v += __shfl_xor(v, off, 32);
    return v;
}

__global__ void node_dots(const float* __restrict__ H, int N, int D,
                          const float* __restrict__ v1,
                          const float* __restrict__ v2,
                          float* __restrict__ o1, float* __restrict__ o2)
{
    int wid  = (blockIdx.x * blockDim.x + threadIdx.x) >> 5;
    int lane = threadIdx.x & 31;
    if (wid >= N) return;
    const float* h = H + (size_t)wid * D;
    float s1 = 0.f, s2 = 0.f;
    for (int i = lane; i < D; i += 32) {
        float x = h[i];
        s1 += x * v1[i];
        s2 += x * v2[i];
    }
    s1 = wave_sum(s1);
    s2 = wave_sum(s2);
    if (lane == 0) { o1[wid] = s1; o2[wid] = s2; }
}

// ---------------------------------------------------------------------------
// Monotonic float<->uint encoding for atomicMax-based segment max.
// ---------------------------------------------------------------------------
__device__ __forceinline__ unsigned fenc(float f)
{
    unsigned u = __float_as_uint(f);
    return (u & 0x80000000u) ? ~u : (u | 0x80000000u);
}
__device__ __forceinline__ float fdec(unsigned e)
{
    return (e & 0x80000000u) ? __uint_as_float(e & 0x7FFFFFFFu)
                             : __uint_as_float(~e);
}

// Pass 1: e = leaky_relu(as[src]+ad[dst], 0.2); segment max over dst.
__global__ void edge_logits(const int* __restrict__ ei, int E, int Nself,
                            const float* __restrict__ asrc,
                            const float* __restrict__ adst,
                            float* __restrict__ ebuf,
                            unsigned* __restrict__ menc)
{
    int e = blockIdx.x * blockDim.x + threadIdx.x;
    if (e >= E + Nself) return;
    int s, d;
    if (e < E) { s = ei[e]; d = ei[E + e]; } else { s = d = e - E; }
    float v = asrc[s] + adst[d];
    v = (v >= 0.f) ? v : 0.2f * v;
    ebuf[e] = v;
    atomicMax(&menc[d], fenc(v));
}

// Pass 2: ex = exp(e - m[dst]); segment sum over dst.
__global__ void edge_exp(const int* __restrict__ ei, int E, int Nself,
                         const float* __restrict__ ebuf,
                         const unsigned* __restrict__ menc,
                         float* __restrict__ exbuf,
                         float* __restrict__ denom)
{
    int e = blockIdx.x * blockDim.x + threadIdx.x;
    if (e >= E + Nself) return;
    int d = (e < E) ? ei[E + e] : (e - E);
    float ex = expf(ebuf[e] - fdec(menc[d]));
    exbuf[e] = ex;
    atomicAdd(&denom[d], ex);
}

// Pass 3: acc[dst,:] += (ex/denom[dst]) * H[src,:]  (float4 per thread)
__global__ void edge_scatter(const int* __restrict__ ei, int E, int Nself, int D,
                             const float* __restrict__ exbuf,
                             const float* __restrict__ denom,
                             const float* __restrict__ H,
                             float* __restrict__ acc)
{
    int g = D >> 2;                                   // float4 groups per edge
    long long t = (long long)blockIdx.x * blockDim.x + threadIdx.x;
    if (t >= (long long)(E + Nself) * g) return;
    int e = (int)(t / g);
    int f = (int)(t % g) << 2;
    int s, d;
    if (e < E) { s = ei[e]; d = ei[E + e]; } else { s = d = e - E; }
    float alpha = exbuf[e] / denom[d];
    float4 h = *(const float4*)(H + (size_t)s * D + f);
    float* o = acc + (size_t)d * D + f;
    atomicAdd(o + 0, alpha * h.x);
    atomicAdd(o + 1, alpha * h.y);
    atomicAdd(o + 2, alpha * h.z);
    atomicAdd(o + 3, alpha * h.w);
}

// x[i,j] += bias[j]; optional relu
__global__ void bias_act(float* __restrict__ x, const float* __restrict__ bias,
                         int N, int D, int do_relu)
{
    long long i = (long long)blockIdx.x * blockDim.x + threadIdx.x;
    if (i >= (long long)N * D) return;
    float v = x[i] + bias[(int)(i % D)];
    x[i] = do_relu ? fmaxf(v, 0.f) : v;
}

// out[k] = sigmoid(s[mask[k,0]] + t[mask[k,1]] + b_out)
__global__ void link_head(const int* __restrict__ mask, int NPAIR,
                          const float* __restrict__ s,
                          const float* __restrict__ t,
                          const float* __restrict__ bout,
                          float* __restrict__ out)
{
    int k = blockIdx.x * blockDim.x + threadIdx.x;
    if (k >= NPAIR) return;
    float z = s[mask[2 * k]] + t[mask[2 * k + 1]] + bout[0];
    out[k] = 1.f / (1.f + expf(-z));
}

static inline int cdiv(long long a, int b) { return (int)((a + b - 1) / b); }

extern "C" void kernel_launch(void* const* d_in, const int* in_sizes, int n_in,
                              void* d_out, int out_size, void* d_ws, size_t ws_size,
                              hipStream_t stream)
{
    (void)n_in; (void)ws_size; (void)out_size;
    const float* xp     = (const float*)d_in[0];
    const int*   ei_ppi = (const int*)  d_in[5];
    const int*   mask   = (const int*)  d_in[6];
    const float* Wa2    = (const float*)d_in[7]  + 2 * FDIM * HDIM;  // Wa[2]
    const float* asa2   = (const float*)d_in[8]  + 2 * HDIM;
    const float* ada2   = (const float*)d_in[9]  + 2 * HDIM;
    const float* ba2    = (const float*)d_in[10] + 2 * HDIM;
    const float* Wb2    = (const float*)d_in[11] + 2 * HDIM * FDIM;  // Wb[2]
    const float* asb2   = (const float*)d_in[12] + 2 * FDIM;
    const float* adb2   = (const float*)d_in[13] + 2 * FDIM;
    const float* bb2    = (const float*)d_in[14] + 2 * FDIM;
    const float* Wout   = (const float*)d_in[21];                    // [2F,1]
    const float* bout   = (const float*)d_in[22];
    float*       out    = (float*)d_out;

    const int NP    = in_sizes[0] / FDIM;      // 50000
    const int E     = in_sizes[5] / 2;         // 800000
    const int NPAIR = in_sizes[6] / 2;         // 200000
    const int ET    = E + NP;                  // edges + self loops

    // -------- workspace layout (peak ~110 MB) --------
    char* ws = (char*)d_ws;
    const size_t szH1 = (size_t)NP * HDIM * 4;       // 51.2 MB
    const size_t szH2 = (size_t)NP * FDIM * 4;       // 25.6 MB
    float*    H1    = (float*)(ws);                  // layer-a features
    float*    H2    = (float*)(ws);                  // reuses H1 region
    float*    accB  = (float*)(ws + szH2);           // hp2, within old H1 region
    float*    accA  = (float*)(ws + szH1);           // hp
    char*     p     = ws + 2 * szH1;
    float*    ebuf  = (float*)p;            p += (size_t)ET * 4;
    float*    exbuf = (float*)p;            p += (size_t)ET * 4;
    unsigned* menc  = (unsigned*)p;         p += (size_t)NP * 4;
    float*    denom = (float*)p;            p += (size_t)NP * 4;
    float*    n1    = (float*)p;            p += (size_t)NP * 4;
    float*    n2    = (float*)p;

    const int TB = 256;

    // ================= layer a (ppi GAT, D = 256) =================
    gemm_wmma_f16<FDIM><<<dim3(cdiv(NP, 128), HDIM / 16), 256, 0, stream>>>(
        xp, Wa2, H1, NP, HDIM);
    node_dots<<<cdiv((long long)NP * 32, TB), TB, 0, stream>>>(
        H1, NP, HDIM, asa2, ada2, n1, n2);

    hipMemsetAsync(menc,  0, (size_t)NP * 4, stream);
    hipMemsetAsync(denom, 0, (size_t)NP * 4, stream);
    hipMemsetAsync(accA,  0, szH1, stream);

    edge_logits<<<cdiv(ET, TB), TB, 0, stream>>>(ei_ppi, E, NP, n1, n2, ebuf, menc);
    edge_exp   <<<cdiv(ET, TB), TB, 0, stream>>>(ei_ppi, E, NP, ebuf, menc, exbuf, denom);
    edge_scatter<<<cdiv((long long)ET * (HDIM / 4), TB), TB, 0, stream>>>(
        ei_ppi, E, NP, HDIM, exbuf, denom, H1, accA);
    bias_act<<<cdiv((long long)NP * HDIM, TB), TB, 0, stream>>>(
        accA, ba2, NP, HDIM, 1);                      // accA = hp = relu(.)

    // ================= layer b (ppi GAT, D = 128) =================
    gemm_wmma_f16<HDIM><<<dim3(cdiv(NP, 128), FDIM / 16), 256, 0, stream>>>(
        accA, Wb2, H2, NP, FDIM);
    node_dots<<<cdiv((long long)NP * 32, TB), TB, 0, stream>>>(
        H2, NP, FDIM, asb2, adb2, n1, n2);

    hipMemsetAsync(menc,  0, (size_t)NP * 4, stream);
    hipMemsetAsync(denom, 0, (size_t)NP * 4, stream);
    hipMemsetAsync(accB,  0, szH2, stream);

    edge_logits<<<cdiv(ET, TB), TB, 0, stream>>>(ei_ppi, E, NP, n1, n2, ebuf, menc);
    edge_exp   <<<cdiv(ET, TB), TB, 0, stream>>>(ei_ppi, E, NP, ebuf, menc, exbuf, denom);
    edge_scatter<<<cdiv((long long)ET * (FDIM / 4), TB), TB, 0, stream>>>(
        ei_ppi, E, NP, FDIM, exbuf, denom, H2, accB);
    bias_act<<<cdiv((long long)NP * FDIM, TB), TB, 0, stream>>>(
        accB, bb2, NP, FDIM, 0);                      // accB = hp2

    // ================= link-prediction head =================
    node_dots<<<cdiv((long long)NP * 32, TB), TB, 0, stream>>>(
        accB, NP, FDIM, Wout, Wout + FDIM, n1, n2);   // s, t
    link_head<<<cdiv(NPAIR, TB), TB, 0, stream>>>(mask, NPAIR, n1, n2, bout, out);
}
